// DotsGenerator_57629871178362
// MI455X (gfx1250) — compile-verified, matrix operand order checked by reading, not run on hardware
//
#include <hip/hip_runtime.h>
#include <hip/hip_bf16.h>

typedef __attribute__((ext_vector_type(16))) _Float16 v16h;
typedef __attribute__((ext_vector_type(8)))  _Float16 v8h;
typedef __attribute__((ext_vector_type(8)))  float    v8f;

#define IMG_H 1080
#define IMG_W 1920
#define IMG_PLANE (IMG_H * IMG_W)
#define NCROP 512
#define CROPSZ 40
#define PADW 42
#define NPIX 1600
#define OCP 64          // padded channels (51 -> 64)
#define K3 81600        // conv3 K = 51*40*40 (exact multiple of 32)

// LDS layout offsets (bytes), all 32B aligned
#define OFF_IN    0
#define OFF_ACT1  10592                       // [42*42][64] f16 = 225792 B
#define OFF_W1F   (10592 + 225792)            // 4 frags * 512 halves = 4096 B
#define OFF_W2F   (10592 + 225792 + 4096)     // 72 frags * 512 halves = 73728 B
#define SMEM_TOTAL (10592 + 225792 + 4096 + 73728)   // 314208 B < 320 KB

__constant__ int c_dot[17][2] = {
  {30,20},{20,30},{10,20},{20,10},{40,20},{34,34},{20,40},{6,34},
  {0,20},{6,6},{20,0},{34,6},{17,20},{23,20},{20,17},{20,23},{20,20}};

// ---------------------------------------------------------------------------
// Fused crop + conv1(3x3,pad1)+BN+relu + conv2(3x3,pad1)+BN+relu.
// One workgroup (8 wave32) per crop.  conv2's K dimension is ordered
// j*64+ic so every 16-wide K-half sits in one spatial tap: A fragments are
// single contiguous 32B LDS vector loads.  W1/W2 are pre-swizzled in LDS
// into per-lane fragment order so B fragments are contiguous loads too.
// conv2 output -> A3[crop][ic*1600+pixel] f16 (conv3's A matrix), packed
// 16B stores.
// ---------------------------------------------------------------------------
__global__ __launch_bounds__(256) void conv12_kernel(
    const float* __restrict__ img, const int* __restrict__ targets,
    const float* __restrict__ w1, const float* __restrict__ g1,
    const float* __restrict__ b1, const float* __restrict__ m1,
    const float* __restrict__ v1,
    const float* __restrict__ w2, const float* __restrict__ g2,
    const float* __restrict__ b2, const float* __restrict__ m2,
    const float* __restrict__ v2,
    _Float16* __restrict__ A3)
{
  extern __shared__ char smem[];
  _Float16* s_in   = (_Float16*)(smem + OFF_IN);    // [3][42][42]
  _Float16* s_act1 = (_Float16*)(smem + OFF_ACT1);  // [42*42][64]
  _Float16* s_w1f  = (_Float16*)(smem + OFF_W1F);   // [4 frag][32 lane][16]
  _Float16* s_w2f  = (_Float16*)(smem + OFF_W2F);   // [72 frag][32 lane][16]

  const int tid   = threadIdx.x;
  const int lane  = tid & 31;
  const int wave  = tid >> 5;
  const int crop  = blockIdx.x;
  const int y0    = targets[crop*5 + 0];
  const int x0    = targets[crop*5 + 1];
  const int lcol  = lane & 15;               // N column / A row owned by lane
  const int khalf = (lane < 16) ? 0 : 16;    // K half owned by lane
  const int mbase = (lane < 16) ? 0 : 8;     // C-matrix row base owned by lane

  // Zero the padded input and padded act1 (borders + channels 51..63 stay 0).
  { uint32_t* z = (uint32_t*)s_in;
    for (int i = tid; i < 2646; i += 256) z[i] = 0u; }
  { uint4* z = (uint4*)s_act1;               // 225792/16 = 14112 b128 stores
    uint4 zero = {0u,0u,0u,0u};
    for (int i = tid; i < 14112; i += 256) z[i] = zero; }

  // Crop -> f16 into padded LDS tile.
  for (int i = tid; i < 3*NPIX; i += 256) {
    int c = i / NPIX, r = i - c*NPIX;
    int yy = r / CROPSZ, xx = r - yy*CROPSZ;
    float px = img[(size_t)c*IMG_PLANE + (size_t)(y0+yy)*IMG_W + (x0+xx)];
    s_in[(c*PADW + yy + 1)*PADW + (xx + 1)] = (_Float16)px;
  }
  // W1 fragments: frag nt, per-lane 16 halves; K order k = ic*9+j (27->32).
  for (int i = tid; i < 4*512; i += 256) {
    int h = i & 15, l = (i >> 4) & 31, nt = i >> 9;
    int k  = ((l < 16) ? 0 : 16) + h;
    int oc = nt*16 + (l & 15);
    float w = (k < 27 && oc < 51) ? w1[oc*27 + k] : 0.f;
    s_w1f[i] = (_Float16)w;
  }
  // W2 fragments: frag = kc*4+nt (kc<18); K order k = j*64+ic.
  for (int i = tid; i < 72*512; i += 256) {
    int h = i & 15, l = (i >> 4) & 31, f = i >> 9;
    int kc = f >> 2, nt = f & 3;
    int j  = kc >> 1;
    int ic = ((kc & 1) << 5) + ((l < 16) ? 0 : 16) + h;
    int oc = nt*16 + (l & 15);
    float w = (ic < 51 && oc < 51) ? w2[oc*459 + ic*9 + j] : 0.f;
    s_w2f[i] = (_Float16)w;
  }
  __syncthreads();

  // ---------------- conv1: C[1600,64] = A[1600,32] x B[32,64] --------------
  for (int mt = wave; mt < 100; mt += 8) {
    const int p  = mt*16 + lcol;
    const int py = p / CROPSZ, px = p - py*CROPSZ;
    v16h a;
#pragma unroll
    for (int h = 0; h < 16; ++h) {
      int k = khalf + h;
      _Float16 val = (_Float16)0.f;
      if (k < 27) {
        int c = k / 9, j = k - c*9;
        int dy = j/3 - 1, dx = j - (j/3)*3 - 1;
        val = s_in[(c*PADW + py + 1 + dy)*PADW + (px + 1 + dx)];
      }
      a[h] = val;
    }
    // Batch the 4 independent WMMAs first (covers WMMA->VALU hazard windows),
    // then unpack.
    v8f acc[4];
#pragma unroll
    for (int nt = 0; nt < 4; ++nt) {
      v16h b = *(const v16h*)(s_w1f + nt*512 + lane*16);
      v8f z = {};
      acc[nt] = __builtin_amdgcn_wmma_f32_16x16x32_f16(false, a, false, b,
                                                       (short)0, z, false, false);
    }
    // 8 unpacked rows = 8 consecutive pixels; 8 | 40 so the image row is
    // constant across them -> one division per tile.
    const int pp0 = mt*16 + mbase;
    const int yy0 = pp0 / CROPSZ, xx0 = pp0 - yy0*CROPSZ;
    const int abase = ((yy0+1)*PADW + (xx0+1))*OCP;
#pragma unroll
    for (int nt = 0; nt < 4; ++nt) {
      const int oc = nt*16 + lcol;
      if (oc < 51) {
        float sc = g1[oc] * rsqrtf(v1[oc] + 1e-5f);
        float sh = b1[oc] - m1[oc]*sc;
#pragma unroll
        for (int vv = 0; vv < 8; ++vv) {
          float r = fmaxf(acc[nt][vv]*sc + sh, 0.f);
          s_act1[abase + vv*OCP + oc] = (_Float16)r;
        }
      }
    }
  }
  __syncthreads();

  // -------- conv2: C[1600,64] = A[1600,576] x B[576,64], K = j*64+ic -------
  // mtile pairs: each B fragment feeds 2 WMMAs; A fragments are single
  // contiguous 32B LDS loads (ds_load_b128).
  for (int mp = wave; mp < 50; mp += 8) {
    const int pA  = (mp*2    )*16 + lcol;
    const int pB  = (mp*2 + 1)*16 + lcol;
    const int pyA = pA / CROPSZ, pxA = pA - pyA*CROPSZ;
    const int pyB = pB / CROPSZ, pxB = pB - pyB*CROPSZ;
    v8f acc[2][4] = {};
    for (int j = 0; j < 9; ++j) {
      const int dy = j/3 - 1, dx = j - (j/3)*3 - 1;
      const _Float16* rowA = s_act1 + ((pyA+1+dy)*PADW + (pxA+1+dx))*OCP;
      const _Float16* rowB = s_act1 + ((pyB+1+dy)*PADW + (pxB+1+dx))*OCP;
#pragma unroll
      for (int hk = 0; hk < 2; ++hk) {             // two 32-wide K chunks / tap
        const int kc  = j*2 + hk;
        const int icb = hk*32 + khalf;
        v16h aA = *(const v16h*)(rowA + icb);
        v16h aB = *(const v16h*)(rowB + icb);
#pragma unroll
        for (int nt = 0; nt < 4; ++nt) {
          v16h b = *(const v16h*)(s_w2f + (kc*4 + nt)*512 + lane*16);
          acc[0][nt] = __builtin_amdgcn_wmma_f32_16x16x32_f16(
              false, aA, false, b, (short)0, acc[0][nt], false, false);
          acc[1][nt] = __builtin_amdgcn_wmma_f32_16x16x32_f16(
              false, aB, false, b, (short)0, acc[1][nt], false, false);
        }
      }
    }
    // BN + relu, pack 8 consecutive pixels -> one 16B store per (mt, nt).
#pragma unroll
    for (int half = 0; half < 2; ++half) {
      const int mt = mp*2 + half;
#pragma unroll
      for (int nt = 0; nt < 4; ++nt) {
        const int oc = nt*16 + lcol;
        if (oc < 51) {
          float sc = g2[oc]*rsqrtf(v2[oc] + 1e-5f);
          float sh = b2[oc] - m2[oc]*sc;
          v8h pack;
#pragma unroll
          for (int vv = 0; vv < 8; ++vv)
            pack[vv] = (_Float16)fmaxf(acc[half][nt][vv]*sc + sh, 0.f);
          *(v8h*)(A3 + (size_t)crop*K3 + (size_t)oc*NPIX + mt*16 + mbase) = pack;
        }
      }
    }
  }
}

// ---------------------------------------------------------------------------
// w3 (51,51,40,40) f32 -> B3[oc (pad 64)][K3] f16, K-contiguous.
// ---------------------------------------------------------------------------
__global__ __launch_bounds__(256) void prep_w3_kernel(
    const float* __restrict__ w3, _Float16* __restrict__ B3)
{
  int i = blockIdx.x*256 + threadIdx.x;       // over 64*81600
  if (i >= OCP*K3) return;
  int oc = i / K3, k = i - oc*K3;
  float w = (oc < 51) ? w3[(size_t)oc*K3 + k] : 0.f;
  B3[i] = (_Float16)w;
}

// ---------------------------------------------------------------------------
// conv3 as GEMM: C[512,64] = A3[512,81600] x B3[64,81600]^T.
// 32 WGs x 8 waves split K (2550 chunks of 32); aligned v16h global loads;
// A stream (HBM-resident, 83 MB) prefetched ahead; partials reduced through
// LDS; BN + relu + clip(0,255).
// ---------------------------------------------------------------------------
__global__ __launch_bounds__(256) void conv3_kernel(
    const _Float16* __restrict__ A3, const _Float16* __restrict__ B3,
    const float* __restrict__ g3, const float* __restrict__ b3,
    const float* __restrict__ m3, const float* __restrict__ v3,
    float* __restrict__ vals)
{
  __shared__ float s_red[8][16*OCP];          // 32 KB
  const int tid   = threadIdx.x;
  const int lane  = tid & 31;
  const int wave  = tid >> 5;
  const int mt    = blockIdx.x;               // 0..31 (16 crops each)
  const int lcol  = lane & 15;
  const int khalf = (lane < 16) ? 0 : 16;
  const int row   = mt*16 + lcol;             // crop index this lane feeds

  v8f acc0 = {}, acc1 = {}, acc2 = {}, acc3 = {};
  for (int kc = wave; kc < 2550; kc += 8) {
    const int kb = kc*32 + khalf;
    const _Float16* aptr = A3 + (size_t)row*K3 + kb;
    // Prefetch this lane's A chunk for the next iteration (global_prefetch_b8).
    __builtin_prefetch(aptr + 8*32, 0, 1);
    v16h a = *(const v16h*)aptr;
    v16h b;
    b = *(const v16h*)(B3 + (size_t)( 0 + lcol)*K3 + kb);
    acc0 = __builtin_amdgcn_wmma_f32_16x16x32_f16(false, a, false, b,
                                                  (short)0, acc0, false, false);
    b = *(const v16h*)(B3 + (size_t)(16 + lcol)*K3 + kb);
    acc1 = __builtin_amdgcn_wmma_f32_16x16x32_f16(false, a, false, b,
                                                  (short)0, acc1, false, false);
    b = *(const v16h*)(B3 + (size_t)(32 + lcol)*K3 + kb);
    acc2 = __builtin_amdgcn_wmma_f32_16x16x32_f16(false, a, false, b,
                                                  (short)0, acc2, false, false);
    b = *(const v16h*)(B3 + (size_t)(48 + lcol)*K3 + kb);
    acc3 = __builtin_amdgcn_wmma_f32_16x16x32_f16(false, a, false, b,
                                                  (short)0, acc3, false, false);
  }

  const int mbase = (lane < 16) ? 0 : 8;
  v8f accs[4] = {acc0, acc1, acc2, acc3};
#pragma unroll
  for (int nt = 0; nt < 4; ++nt)
#pragma unroll
    for (int vv = 0; vv < 8; ++vv)
      s_red[wave][(mbase + vv)*OCP + nt*16 + lcol] = accs[nt][vv];
  __syncthreads();

  for (int e = tid; e < 16*OCP; e += 256) {
    float s = 0.f;
#pragma unroll
    for (int w = 0; w < 8; ++w) s += s_red[w][e];
    int m = e >> 6, oc = e & 63;
    float out = 0.f;
    if (oc < 51) {
      float sc = g3[oc]*rsqrtf(v3[oc] + 1e-5f);
      float sh = b3[oc] - m3[oc]*sc;
      out = fminf(fmaxf(s*sc + sh, 0.f), 255.f);
    }
    vals[(size_t)(mt*16 + m)*OCP + oc] = out;
  }
}

// ---------------------------------------------------------------------------
// Stamp 512 crops x 17 dots x 9 neighbors into the output image (CHW).
// ---------------------------------------------------------------------------
__global__ __launch_bounds__(256) void scatter_kernel(
    const int* __restrict__ targets, const float* __restrict__ vals,
    float* __restrict__ out)
{
  int i = blockIdx.x*256 + threadIdx.x;       // over 512*17*9
  if (i >= NCROP*17*9) return;
  int crop = i / 153;
  int rem  = i - crop*153;
  int d    = rem / 9;
  int r    = rem - d*9;
  int dy = r/3 - 1, dx = r - (r/3)*3 - 1;
  int y = targets[crop*5 + 0] + c_dot[d][0] + dy;
  int x = targets[crop*5 + 1] + c_dot[d][1] + dx;
  if ((unsigned)y < IMG_H && (unsigned)x < IMG_W) {
#pragma unroll
    for (int c = 0; c < 3; ++c)
      out[(size_t)c*IMG_PLANE + (size_t)y*IMG_W + x] =
          vals[(size_t)crop*OCP + d*3 + c];
  }
}

extern "C" void kernel_launch(void* const* d_in, const int* in_sizes, int n_in,
                              void* d_out, int out_size, void* d_ws, size_t ws_size,
                              hipStream_t stream)
{
  const float* img     = (const float*)d_in[0];
  const int*   targets = (const int*)  d_in[1];
  const float* w1 = (const float*)d_in[2];
  const float* g1 = (const float*)d_in[3];
  const float* b1 = (const float*)d_in[4];
  const float* m1 = (const float*)d_in[5];
  const float* v1 = (const float*)d_in[6];
  const float* w2 = (const float*)d_in[7];
  const float* g2 = (const float*)d_in[8];
  const float* b2 = (const float*)d_in[9];
  const float* m2 = (const float*)d_in[10];
  const float* v2 = (const float*)d_in[11];
  const float* w3 = (const float*)d_in[12];
  const float* g3 = (const float*)d_in[13];
  const float* b3 = (const float*)d_in[14];
  const float* m3 = (const float*)d_in[15];
  const float* v3 = (const float*)d_in[16];
  float* out = (float*)d_out;

  // Workspace layout (all 32B-aligned):
  char* ws = (char*)d_ws;
  _Float16* A3   = (_Float16*)ws;                                   // 512*81600 f16
  _Float16* B3   = (_Float16*)(ws + (size_t)NCROP*K3*2);            //  64*81600 f16
  float*    vals = (float*)   (ws + (size_t)NCROP*K3*2 + (size_t)OCP*K3*2); // 512*64 f32

  // Background image copy (d2d async, graph-capture safe).
  hipMemcpyAsync(d_out, d_in[0], (size_t)3*IMG_PLANE*sizeof(float),
                 hipMemcpyDeviceToDevice, stream);

  prep_w3_kernel<<<(OCP*K3 + 255)/256, 256, 0, stream>>>(w3, B3);

  hipFuncSetAttribute((const void*)conv12_kernel,
                      hipFuncAttributeMaxDynamicSharedMemorySize, (int)SMEM_TOTAL);
  conv12_kernel<<<NCROP, 256, SMEM_TOTAL, stream>>>(img, targets,
                                                    w1, g1, b1, m1, v1,
                                                    w2, g2, b2, m2, v2, A3);

  conv3_kernel<<<32, 256, 0, stream>>>(A3, B3, g3, b3, m3, v3, vals);

  scatter_kernel<<<(NCROP*17*9 + 255)/256, 256, 0, stream>>>(targets, vals, out);
}
